// GIN_5995774345340
// MI455X (gfx1250) — matrix operand order repaired and checked by reading, not compiled, not run on hardware
//
#include <hip/hip_runtime.h>

#define NN 50000
#define GD 256
#define GL 4
#define NE 800000
#define LD_OUT (GL*GD)   // 1024 output row stride
#define PADH 264         // LDS row stride in halves (256 + 8 pad) = 528 B = 132 dwords

typedef __attribute__((ext_vector_type(16))) _Float16 v16h;
typedef __attribute__((ext_vector_type(8)))  _Float16 v8h;
typedef __attribute__((ext_vector_type(4)))  _Float16 v4h;
typedef __attribute__((ext_vector_type(8)))  float    v8f;
typedef __attribute__((ext_vector_type(4)))  float    v4f;

typedef __attribute__((address_space(3))) _Float16 lds_h;   // 32-bit LDS pointers
typedef __attribute__((address_space(3))) const v8h lds_v8h;

// ---- W [256x256] f32 row-major (k,n) -> Wt f16 (n,k) ----
__global__ void k_transpose_w(const float* __restrict__ W, _Float16* __restrict__ Wt) {
    int i = blockIdx.x * blockDim.x + threadIdx.x;   // 65536
    int k = i >> 8, n = i & 255;
    Wt[n * GD + k] = (_Float16)W[k * GD + n];
}

// ---- agg[dst] += h[src]  (one edge per block, one column per thread) ----
__global__ void k_scatter_add(const float* __restrict__ h, int hstride,
                              const long long* __restrict__ srci,
                              const long long* __restrict__ dsti,
                              float* __restrict__ agg) {
    int e = blockIdx.x;
    int d = threadIdx.x;
    long long s = srci[e], t = dsti[e];
    atomicAdd(&agg[(size_t)t * GD + d], h[(size_t)s * hstride + d]);
}

// ---- r = h + agg, cast to f16 GEMM input (4 elems/thread) ----
__global__ void k_make_r16(const float* __restrict__ h, int hstride,
                           const float* __restrict__ agg, _Float16* __restrict__ x16) {
    int c = blockIdx.x * blockDim.x + threadIdx.x;   // NN*GD/4 chunks
    int row = c >> 6, col4 = (c & 63) * 4;
    v4f hv = *(const v4f*)(h + (size_t)row * hstride + col4);
    v4f av = *(const v4f*)(agg + (size_t)row * GD + col4);
    v4h o;
    o.x = (_Float16)(hv.x + av.x); o.y = (_Float16)(hv.y + av.y);
    o.z = (_Float16)(hv.z + av.z); o.w = (_Float16)(hv.w + av.w);
    *(v4h*)(x16 + (size_t)row * GD + col4) = o;
}

// ---- Y = X @ W + bias (f16 WMMA, f32 accum); fused per-column sum/sumsq for BN ----
// Wt staged in LDS via async global->LDS copy; one wave = 16-row x 256-col strip.
__global__ void __launch_bounds__(256, 2)
k_gemm_f16(const _Float16* __restrict__ X, const _Float16* __restrict__ Wt,
           const float* __restrict__ bias, float* __restrict__ Y,
           float* __restrict__ colsum, float* __restrict__ colsumsq, int nrows) {
    __shared__ _Float16 sw[GD * PADH];               // 132 KB, padded rows

    // Stage Wt (n-major, 256x256 f16) into LDS with async copies (ASYNCcnt-tracked).
    // 8192 chunks of 16 B; 32 chunks per thread.
    for (int c = threadIdx.x; c < 8192; c += 256) {
        int n  = c >> 5;
        int kc = (c & 31) * 8;
        unsigned           ldst = (unsigned)(size_t)(&sw[n * PADH + kc]);
        unsigned long long gsrc = (unsigned long long)(size_t)(Wt + n * GD + kc);
        asm volatile("global_load_async_to_lds_b128 %0, %1, off"
                     :: "v"(ldst), "v"(gsrc) : "memory");
    }
    asm volatile("s_wait_asynccnt 0" ::: "memory");
    __syncthreads();

    int wave  = threadIdx.x >> 5;
    int lane  = threadIdx.x & 31;
    int strip = blockIdx.x * 8 + wave;
    int row0  = strip * 16;
    if (row0 >= nrows) return;             // wave-uniform exit (EXEC all-1 for WMMA)

    int mrow  = lane & 15;                 // A: M row / B: N column / C: column
    int khalf = lane >> 4;                 // lane-group selector

    const _Float16* xrow = X + (size_t)(row0 + mrow) * GD;

    v8f acc[16];
    v8f z = {0.f, 0.f, 0.f, 0.f, 0.f, 0.f, 0.f, 0.f};
#pragma unroll
    for (int t = 0; t < 16; t++) acc[t] = z;

    // Two LDS base registers (32-bit, addrspace(3)); the asm barrier keeps the
    // compiler from folding b1 back into b0 + >64K constants, so every DS access
    // in the hot loop is a compile-time immediate < 64 KB off b0 (tiles 0-7) or
    // b1 (tiles 8-15): no address VALU, no WMMA WAR-hazard NOPs.
    const lds_h* b0 = (const lds_h*)(unsigned)(size_t)(&sw[mrow * PADH + khalf * 8]);
    const lds_h* b1 = b0 + 8 * 16 * PADH;
    asm volatile("" : "+v"(b1));

    // A 16x32 f16 layout: lane holds K = kb + khalf*8 + {0..7} and kb+16+khalf*8 + {0..7}
    union AU { v16h v; v8h p[2]; };
    AU a;
    a.p[0] = *(const v8h*)(xrow + khalf * 8);
    a.p[1] = *(const v8h*)(xrow + 16 + khalf * 8);

#pragma unroll
    for (int kb = 0; kb < GD; kb += 32) {
        AU cur = a;
        if (kb + 32 < GD) {                // prefetch next k-step's A tile (global)
            a.p[0] = *(const v8h*)(xrow + kb + 32 + khalf * 8);
            a.p[1] = *(const v8h*)(xrow + kb + 48 + khalf * 8);
        }
        // B 32x16 f16: lane = column n = t*16+mrow, K = kb + khalf*8 + {0..7} and kb+16+khalf*8 + {0..7}
        // Double-buffered LDS reads: load tile t+1 while WMMA consumes tile t.
        union BU { v16h v; v8h p[2]; };
        BU bb[2];
        bb[0].p[0] = *(lds_v8h*)(b0 + kb);
        bb[0].p[1] = *(lds_v8h*)(b0 + kb + 16);
#pragma unroll
        for (int t = 0; t < 16; t++) {
            if (t < 15) {
                int tn = t + 1;
                const lds_h* bp = (tn < 8 ? b0 + tn * 16 * PADH
                                          : b1 + (tn - 8) * 16 * PADH) + kb;
                bb[tn & 1].p[0] = *(lds_v8h*)bp;
                bb[tn & 1].p[1] = *(lds_v8h*)(bp + 16);
            }
            acc[t] = __builtin_amdgcn_wmma_f32_16x16x32_f16(
                false, cur.v, false, bb[t & 1].v, (short)0, acc[t], false, false);
        }
    }

    // Epilogue: C/D layout — lane<16: col=lane, M=v ; lane>=16: col=lane-16, M=v+8
#pragma unroll
    for (int t = 0; t < 16; t++) {
        int col = t * 16 + mrow;
        float bv = bias[col];
        float s = 0.f, sq = 0.f;
#pragma unroll
        for (int v = 0; v < 8; v++) {
            float y = acc[t][v] + bv;
            Y[(size_t)(row0 + v + khalf * 8) * GD + col] = y;
            s += y; sq += y * y;
        }
        atomicAdd(&colsum[col], s);
        atomicAdd(&colsumsq[col], sq);
    }
}

// ---- finalize BN affine: scale = g*rsqrt(var+eps), shift = b - mu*scale ----
__global__ void k_finalize_bn(const float* __restrict__ sum, const float* __restrict__ sumsq,
                              const float* __restrict__ g, const float* __restrict__ b,
                              float* __restrict__ scale, float* __restrict__ shift) {
    int c = threadIdx.x;
    float mu  = sum[c] * (1.0f / NN);
    float var = sumsq[c] * (1.0f / NN) - mu * mu;
    float sc  = g[c] * rsqrtf(var + 1e-5f);
    scale[c] = sc;
    shift[c] = b[c] - mu * sc;
}

// ---- x16 = f16(relu(Y*scale + shift)) : BN1 apply, feeds GEMM2 (4 elems/thread) ----
__global__ void k_bn_relu_f16(const float* __restrict__ Y, const float* __restrict__ scale,
                              const float* __restrict__ shift, _Float16* __restrict__ x16) {
    int c = blockIdx.x * blockDim.x + threadIdx.x;
    int col4 = (c & 63) * 4;
    v4f y = *(const v4f*)(Y + (size_t)c * 4);
    v4f sc = *(const v4f*)(scale + col4);
    v4f sh = *(const v4f*)(shift + col4);
    v4h o;
    o.x = (_Float16)fmaxf(y.x * sc.x + sh.x, 0.f);
    o.y = (_Float16)fmaxf(y.y * sc.y + sh.y, 0.f);
    o.z = (_Float16)fmaxf(y.z * sc.z + sh.z, 0.f);
    o.w = (_Float16)fmaxf(y.w * sc.w + sh.w, 0.f);
    *(v4h*)(x16 + (size_t)c * 4) = o;
}

// ---- M = relu(Y*scale2 + shift2) (f32) + accumulate BN3 column stats ----
__global__ void k_bn_relu_stats(const float* __restrict__ Y, const float* __restrict__ scale,
                                const float* __restrict__ shift, float* __restrict__ M,
                                float* __restrict__ colsum, float* __restrict__ colsumsq,
                                int rows_per_block) {
    int col = threadIdx.x;
    float sc = scale[col], sh = shift[col];
    int r0 = blockIdx.x * rows_per_block;
    int r1 = r0 + rows_per_block; if (r1 > NN) r1 = NN;
    float s = 0.f, sq = 0.f;
    for (int r = r0; r < r1; ++r) {
        float m = fmaxf(Y[(size_t)r * GD + col] * sc + sh, 0.0f);
        M[(size_t)r * GD + col] = m;
        s += m; sq += m * m;
    }
    atomicAdd(&colsum[col], s);
    atomicAdd(&colsumsq[col], sq);
}

// ---- h_next = relu(M*scale3 + shift3) -> d_out slice (row stride 1024) ----
__global__ void k_bn_relu_out(const float* __restrict__ M, const float* __restrict__ scale,
                              const float* __restrict__ shift, float* __restrict__ out) {
    int c = blockIdx.x * blockDim.x + threadIdx.x;
    int row = c >> 6, col4 = (c & 63) * 4;
    v4f m = *(const v4f*)(M + (size_t)row * GD + col4);
    v4f sc = *(const v4f*)(scale + col4);
    v4f sh = *(const v4f*)(shift + col4);
    v4f o;
    o.x = fmaxf(m.x * sc.x + sh.x, 0.f);
    o.y = fmaxf(m.y * sc.y + sh.y, 0.f);
    o.z = fmaxf(m.z * sc.z + sh.z, 0.f);
    o.w = fmaxf(m.w * sc.w + sh.w, 0.f);
    *(v4f*)(out + (size_t)row * LD_OUT + col4) = o;
}

extern "C" void kernel_launch(void* const* d_in, const int* in_sizes, int n_in,
                              void* d_out, int out_size, void* d_ws, size_t ws_size,
                              hipStream_t stream) {
    (void)in_sizes; (void)n_in; (void)out_size; (void)ws_size;
    const float*      h0   = (const float*)d_in[0];
    const long long*  ei   = (const long long*)d_in[1];   // int64 [2, NE]
    const float*      W1   = (const float*)d_in[2];
    const float*      b1   = (const float*)d_in[3];
    const float*      W2   = (const float*)d_in[4];
    const float*      b2   = (const float*)d_in[5];
    const float*      bn1g = (const float*)d_in[6];
    const float*      bn1b = (const float*)d_in[7];
    const float*      bn2g = (const float*)d_in[8];
    const float*      bn2b = (const float*)d_in[9];
    const float*      bn3g = (const float*)d_in[10];
    const float*      bn3b = (const float*)d_in[11];
    float* out = (float*)d_out;

    // workspace layout
    char* ws = (char*)d_ws;
    size_t off = 0;
    auto alloc = [&](size_t bytes) -> void* {
        void* p = ws + off; off += (bytes + 255) & ~(size_t)255; return p;
    };
    float*    AGG  = (float*)alloc((size_t)NN * GD * 4);     // agg, later reused as M
    float*    Y    = (float*)alloc((size_t)NN * GD * 4);     // GEMM output
    _Float16* X16  = (_Float16*)alloc((size_t)NN * GD * 2);  // f16 GEMM A input
    _Float16* Wt   = (_Float16*)alloc((size_t)2 * GL * GD * GD * 2); // f16 transposed weights
    float*    ssum  = (float*)alloc(GD * 4);
    float*    ssq   = (float*)alloc(GD * 4);
    float*    scale = (float*)alloc(GD * 4);
    float*    shift = (float*)alloc(GD * 4);

    const long long* srci = ei;
    const long long* dsti = ei + NE;

    // weights -> f16, transposed. Wt layout: [layer][0:W1t | 1:W2t][256*256]
    for (int l = 0; l < GL; l++) {
        k_transpose_w<<<GD * GD / 256, 256, 0, stream>>>(W1 + (size_t)l * GD * GD,
                                                         Wt + ((size_t)l * 2 + 0) * GD * GD);
        k_transpose_w<<<GD * GD / 256, 256, 0, stream>>>(W2 + (size_t)l * GD * GD,
                                                         Wt + ((size_t)l * 2 + 1) * GD * GD);
    }

    const int chunks = NN * GD / 4;                   // 3.2M 4-elem chunks
    const int gemm_blocks = (((NN + 15) / 16) + 7) / 8;  // 391

    for (int l = 0; l < GL; l++) {
        const float* hin = (l == 0) ? h0 : (out + (size_t)(l - 1) * GD);
        int hstride = (l == 0) ? GD : LD_OUT;

        // agg = segment_sum(h[src], dst); r = h + agg (f16)
        hipMemsetAsync(AGG, 0, (size_t)NN * GD * 4, stream);
        k_scatter_add<<<NE, GD, 0, stream>>>(hin, hstride, srci, dsti, AGG);
        k_make_r16<<<chunks / 256, 256, 0, stream>>>(hin, hstride, AGG, X16);

        // GEMM1 + BN1 stats -> BN1+ReLU -> f16
        hipMemsetAsync(ssum, 0, GD * 4, stream);
        hipMemsetAsync(ssq,  0, GD * 4, stream);
        k_gemm_f16<<<gemm_blocks, 256, 0, stream>>>(X16, Wt + ((size_t)l * 2 + 0) * GD * GD,
                                                    b1 + l * GD, Y, ssum, ssq, NN);
        k_finalize_bn<<<1, GD, 0, stream>>>(ssum, ssq, bn1g + l * GD, bn1b + l * GD, scale, shift);
        k_bn_relu_f16<<<chunks / 256, 256, 0, stream>>>(Y, scale, shift, X16);

        // GEMM2 + BN2 stats
        hipMemsetAsync(ssum, 0, GD * 4, stream);
        hipMemsetAsync(ssq,  0, GD * 4, stream);
        k_gemm_f16<<<gemm_blocks, 256, 0, stream>>>(X16, Wt + ((size_t)l * 2 + 1) * GD * GD,
                                                    b2 + l * GD, Y, ssum, ssq, NN);
        k_finalize_bn<<<1, GD, 0, stream>>>(ssum, ssq, bn2g + l * GD, bn2b + l * GD, scale, shift);

        // BN2 apply + ReLU -> M (f32, reuses AGG) with fused BN3 stats
        hipMemsetAsync(ssum, 0, GD * 4, stream);
        hipMemsetAsync(ssq,  0, GD * 4, stream);
        k_bn_relu_stats<<<(NN + 127) / 128, GD, 0, stream>>>(Y, scale, shift, AGG, ssum, ssq, 128);
        k_finalize_bn<<<1, GD, 0, stream>>>(ssum, ssq, bn3g + l * GD, bn3b + l * GD, scale, shift);

        // BN3 apply + ReLU -> output slice (also next layer's h)
        k_bn_relu_out<<<chunks / 256, 256, 0, stream>>>(AGG, scale, shift, out + (size_t)l * GD);
    }
}
